// ConvGRUDeformable_68195490726216
// MI455X (gfx1250) — matrix operand order repaired and checked by reading, not compiled, not run on hardware
//
#include <hip/hip_runtime.h>
#include <hip/hip_bf16.h>

// ---------------- problem constants ----------------
#define H_        100
#define W_        200
#define HW_       20000
#define KK_       9
#define KTOT      4608          // 512 channels * 9 taps, K index = kk*512 + c
#define KSTEPS    144           // KTOT / 32
#define HALO_W    202
#define HALO_H    102
#define HALO_PIX  (HALO_W * HALO_H)
#define MT_PER_ROW 13           // ceil(200/16) tiles per image row
#define MTILES    (H_ * MT_PER_ROW)
#define LDSROW    4632          // f16 elems per LDS A row (4608 + 24 pad -> conflict-free banks)
#define LDS_BYTES (16 * LDSROW * 2)   // 148224 B < 320 KB/WGP

typedef _Float16 half8   __attribute__((ext_vector_type(8)));
typedef _Float16 half16  __attribute__((ext_vector_type(16)));
typedef float    float4v __attribute__((ext_vector_type(4)));
typedef float    float8  __attribute__((ext_vector_type(8)));

__device__ __forceinline__ half16 cat8(half8 lo, half8 hi) {
  return __builtin_shufflevector(lo, hi, 0,1,2,3,4,5,6,7,8,9,10,11,12,13,14,15);
}
__device__ __forceinline__ float8 wmma16(half16 a, half16 b, float8 c) {
  // D = A(16x32 f16) * B(32x16 f16) + C(16x16 f32)
  return __builtin_amdgcn_wmma_f32_16x16x32_f16(false, a, false, b, (short)0, c, false, false);
}

// 8 consecutive pixels per lane -> two b128 stores on the fast path.
__device__ __forceinline__ void store8(float* __restrict__ dst, float8 v, bool full, int rem) {
  if (full) {
    *(float4v*)dst       = __builtin_shufflevector(v, v, 0, 1, 2, 3);
    *(float4v*)(dst + 4) = __builtin_shufflevector(v, v, 4, 5, 6, 7);
  } else {
    #pragma unroll
    for (int j = 0; j < 8; ++j)
      if (j < rem) dst[j] = v[j];
  }
}
__device__ __forceinline__ float8 load8(const float* __restrict__ src, bool full, int rem) {
  float8 v = {};
  if (full) {
    float4v a = *(const float4v*)src;
    float4v b = *(const float4v*)(src + 4);
    v = __builtin_shufflevector(a, b, 0, 1, 2, 3, 4, 5, 6, 7);
  } else {
    #pragma unroll
    for (int j = 0; j < 8; ++j)
      if (j < rem) v[j] = src[j];
  }
  return v;
}

// ---------------- kernel 1: pack h/x into NHWC f16 halo buffers ----------------
__global__ void k_pack_halo(const float* __restrict__ h, const float* __restrict__ x,
                            _Float16* __restrict__ Hh, _Float16* __restrict__ Xh,
                            _Float16* __restrict__ RHh) {
  int idx = blockIdx.x * 256 + threadIdx.x;
  if (idx >= HALO_PIX * 32) return;
  int pix = idx >> 5, l = idx & 31;
  int hy = pix / HALO_W, hx = pix % HALO_W;
  int y = hy - 1, xw = hx - 1;
  int c0 = l * 8;
  half8 hv = {}, xv = {};
  size_t base = (size_t)pix * 256 + c0;
  if (y >= 0 && y < H_ && xw >= 0 && xw < W_) {
    int p = y * W_ + xw;
    #pragma unroll
    for (int j = 0; j < 8; ++j) {
      hv[j] = (_Float16)h[(size_t)(c0 + j) * HW_ + p];
      xv[j] = (_Float16)x[(size_t)(c0 + j) * HW_ + p];
    }
    *(half8*)(Hh + base) = hv;
    *(half8*)(Xh + base) = xv;
  } else {
    *(half8*)(Hh + base)  = hv;      // zeros
    *(half8*)(Xh + base)  = xv;
    *(half8*)(RHh + base) = hv;
  }
}

// ---------------- kernel 2: shuffle f32 weights into WMMA B-fragment order ----------------
// B layout (16x16x32): lane n(0..15) = col n, K {k0..k0+7, k0+16..k0+23}; lane n+16 = K {k0+8..k0+15, k0+24..k0+31}.
// Source weight is (ncols, 512, 9) f32 with K = kk*512 + c.
__global__ void k_pack_w(const float* __restrict__ src0, const float* __restrict__ src1,
                         int split, int ncols, int NT, unsigned int* __restrict__ dst) {
  int idx = blockIdx.x * 256 + threadIdx.x;
  int total = KSTEPS * NT * 32 * 8;
  if (idx >= total) return;
  int d = idx & 7, lane = (idx >> 3) & 31;
  int st = idx >> 8;
  int t = st % NT, s = st / NT;
  int n = t * 16 + (lane & 15);
  int kbase = s * 32 + ((lane >> 4) << 3) + ((d >> 2) << 4) + 2 * (d & 3);
  unsigned int out = 0;
  #pragma unroll
  for (int hh = 0; hh < 2; ++hh) {
    int k = kbase + hh;
    float v = 0.f;
    if (n < ncols && k < KTOT) {
      int kk = k >> 9, c = k & 511;
      const float* src = (n < split) ? src0 : src1;
      int nn = (n < split) ? n : n - split;
      v = src[(size_t)nn * KTOT + c * KK_ + kk];
    }
    union { _Float16 f; unsigned short u; } cv; cv.f = (_Float16)v;
    out |= ((unsigned int)cv.u) << (16 * hh);
  }
  dst[idx] = out;
}

// ---------------- kernel 3: 3x3 offset conv as implicit-im2col WMMA GEMM ----------------
// One wave per 16-pixel M-tile; the wave owns all NT N-tiles, so each A fragment
// feeds NT independent WMMA chains (A traffic /NT, better load/compute overlap).
// A layout per ISA: lanes 0-15 hold K {0..7,16..23}, lanes 16-31 hold K {8..15,24..31}.
template<int NT>
__global__ void k_offconv(const _Float16* __restrict__ A0, const _Float16* __restrict__ A1,
                          const _Float16* __restrict__ wfrag,
                          const float* __restrict__ bias0, const float* __restrict__ bias1,
                          int ncols, int split,
                          float* __restrict__ dst) {
  int gwave = (blockIdx.x * 256 + threadIdx.x) >> 5;
  int lane = threadIdx.x & 31;
  if (gwave >= MTILES) return;
  int row = gwave / MT_PER_ROW, px0 = (gwave % MT_PER_ROW) * 16;
  int m = lane & 15, hi = lane >> 4;
  int px = px0 + m; if (px > W_ - 1) px = W_ - 1;   // clamp partial tile (masked on store)

  const size_t bstep = (size_t)NT * 32 * 16;        // halfs per k-step
  const _Float16* bptr[NT];
  #pragma unroll
  for (int t = 0; t < NT; ++t)
    bptr[t] = wfrag + ((size_t)t * 32 + lane) * 16;

  float8 acc[NT];
  #pragma unroll
  for (int t = 0; t < NT; ++t) acc[t] = (float8){};

  #pragma unroll 1
  for (int kk = 0; kk < 9; ++kk) {
    int ky = kk / 3 - 1, kx = kk % 3 - 1;
    size_t pixoff = ((size_t)(row + 1 + ky) * HALO_W + (px + 1 + kx)) * 256 + hi * 8;
    const _Float16* pa = A0 + pixoff;               // channels 0..255
    #pragma unroll
    for (int cb = 0; cb < 8; ++cb) {
      half16 a = cat8(*(const half8*)pa, *(const half8*)(pa + 16));
      #pragma unroll
      for (int t = 0; t < NT; ++t) {
        acc[t] = wmma16(a, *(const half16*)bptr[t], acc[t]);
        bptr[t] += bstep;
      }
      pa += 32;
    }
    pa = A1 + pixoff;                               // channels 256..511
    #pragma unroll
    for (int cb = 0; cb < 8; ++cb) {
      half16 a = cat8(*(const half8*)pa, *(const half8*)(pa + 16));
      #pragma unroll
      for (int t = 0; t < NT; ++t) {
        acc[t] = wmma16(a, *(const half16*)bptr[t], acc[t]);
        bptr[t] += bstep;
      }
      pa += 32;
    }
  }
  // C layout: VGPR j -> M = j + hi*8, N = lane&15 (+ n-tile); 8 consecutive pixels per lane.
  bool full = (px0 + 16 <= W_);
  int rem = W_ - (px0 + hi * 8);
  if (rem > 8) rem = 8; if (rem < 0) rem = 0;
  int pb = row * W_ + px0 + hi * 8;
  #pragma unroll
  for (int t = 0; t < NT; ++t) {
    int ncol = t * 16 + (lane & 15);
    if (ncol < ncols) {
      float bias = (ncol < split) ? bias0[ncol] : bias1[ncol - split];
      float8 o;
      #pragma unroll
      for (int j = 0; j < 8; ++j) o[j] = acc[t][j] + bias;
      store8(dst + (size_t)ncol * HW_ + pb, o, full, rem);
    }
  }
}

// ---------------- kernel 4: deformable conv (gather -> LDS -> WMMA) + fused epilogue ----------------
// mode 0: z = sigmoid(acc)                 -> fout (256,HW)
// mode 1: r = sigmoid(acc); rhh = f16(r*h) -> RHh halo interior
// mode 2: out = (1-z)*h + z*tanh(acc)      -> fout (d_out, NCHW)
__device__ __forceinline__ half16 gather16(const _Float16* __restrict__ src, int y, int x, int cc) {
  half8 lo = {}, hi = {};
  if (y >= 0 && y < H_ && x >= 0 && x < W_) {
    const _Float16* p = src + ((size_t)(y + 1) * HALO_W + (x + 1)) * 256 + cc;
    lo = *(const half8*)p;
    hi = *(const half8*)(p + 8);
  }
  return cat8(lo, hi);
}

__global__ void __launch_bounds__(256, 2) k_deform(
    const _Float16* __restrict__ A0, const _Float16* __restrict__ A1,
    const float* __restrict__ offs,          // (18, HW) channel-major
    const _Float16* __restrict__ wfrag,      // packed B frags, NT = 16
    const float* __restrict__ zbuf, const float* __restrict__ hbuf,
    float* __restrict__ fout, _Float16* __restrict__ rhh,
    int mode) {
  extern __shared__ _Float16 lds[];          // 16 rows x LDSROW f16
  int tid = threadIdx.x, lane = tid & 31, wv = tid >> 5;
  int mtile = blockIdx.x;
  int row = mtile / MT_PER_ROW, px0 = (mtile % MT_PER_ROW) * 16;

  // ---- phase 1: bilinear-gather the 16 x 4608 patch tile into LDS ----
  for (int task = wv; task < 144; task += 8) {    // task = (pixel m, tap kk)
    int m = task / 9, kk = task % 9;
    int px = px0 + m; if (px > W_ - 1) px = W_ - 1;
    int p = row * W_ + px;
    float oy = offs[(size_t)(2 * kk) * HW_ + p];
    float ox = offs[(size_t)(2 * kk + 1) * HW_ + p];
    float fy = (float)row + (float)(kk / 3 - 1) + oy;
    float fx = (float)px  + (float)(kk % 3 - 1) + ox;
    float y0f = floorf(fy), x0f = floorf(fx);
    float dy = fy - y0f, dx = fx - x0f;
    int y0 = (int)y0f, x0 = (int)x0f;
    float w00 = (1.f - dy) * (1.f - dx), w01 = (1.f - dy) * dx;
    float w10 = dy * (1.f - dx),         w11 = dy * dx;

    int c0 = lane * 16;                           // 16 channels per lane, 512 per wave
    const _Float16* src = (c0 < 256) ? A0 : A1;
    int cc = c0 & 255;
    half16 v00 = gather16(src, y0,     x0,     cc);
    half16 v01 = gather16(src, y0,     x0 + 1, cc);
    half16 v10 = gather16(src, y0 + 1, x0,     cc);
    half16 v11 = gather16(src, y0 + 1, x0 + 1, cc);
    half8 olo, ohi;
    #pragma unroll
    for (int i = 0; i < 8; ++i) {
      float a = w00 * (float)v00[i] + w01 * (float)v01[i] +
                w10 * (float)v10[i] + w11 * (float)v11[i];
      float b = w00 * (float)v00[i + 8] + w01 * (float)v01[i + 8] +
                w10 * (float)v10[i + 8] + w11 * (float)v11[i + 8];
      olo[i] = (_Float16)a;
      ohi[i] = (_Float16)b;
    }
    size_t lo = (size_t)m * LDSROW + kk * 512 + c0;
    *(half8*)(lds + lo)     = olo;
    *(half8*)(lds + lo + 8) = ohi;
  }
  __syncthreads();

  // ---- phase 2: GEMM 16x256 from LDS A-tile; each wave owns 2 adjacent N-tiles ----
  int hi = lane >> 4;
  int t0 = wv * 2;
  const _Float16* aptr = lds + (size_t)(lane & 15) * LDSROW + hi * 8;
  const _Float16* b0p = wfrag + ((size_t)t0 * 32 + lane) * 16;
  const _Float16* b1p = b0p + 32 * 16;            // next n-tile
  const size_t bstep = (size_t)16 * 32 * 16;      // halfs per k-step
  float8 acc0 = {}, acc1 = {};
  #pragma unroll 4
  for (int s = 0; s < KSTEPS; ++s) {
    half16 a = cat8(*(const half8*)aptr, *(const half8*)(aptr + 16));
    half16 b0 = *(const half16*)b0p;
    half16 b1 = *(const half16*)b1p;
    __builtin_prefetch(b0p + bstep, 0, 1);
    acc0 = wmma16(a, b0, acc0);
    acc1 = wmma16(a, b1, acc1);
    aptr += 32; b0p += bstep; b1p += bstep;
  }

  // ---- fused epilogue: lane holds 8 consecutive pixels of one output channel ----
  int nc = lane & 15;
  bool full = (px0 + 16 <= W_);
  int rem = W_ - (px0 + hi * 8);
  if (rem > 8) rem = 8; if (rem < 0) rem = 0;
  int pb = row * W_ + px0 + hi * 8;
  #pragma unroll
  for (int hf = 0; hf < 2; ++hf) {
    float8 acc = hf ? acc1 : acc0;
    int ncol = (t0 + hf) * 16 + nc;
    if (mode == 0) {
      float8 o;
      #pragma unroll
      for (int j = 0; j < 8; ++j) o[j] = 1.f / (1.f + __expf(-acc[j]));
      store8(fout + (size_t)ncol * HW_ + pb, o, full, rem);
    } else if (mode == 1) {
      float8 hv = load8(hbuf + (size_t)ncol * HW_ + pb, full, rem);
      size_t rbase = ((size_t)(row + 1) * HALO_W + (px0 + hi * 8 + 1)) * 256 + ncol;
      #pragma unroll
      for (int j = 0; j < 8; ++j) {
        if (full || j < rem) {
          float r = 1.f / (1.f + __expf(-acc[j]));
          rhh[rbase + (size_t)j * 256] = (_Float16)(r * hv[j]);
        }
      }
    } else {
      float8 zv = load8(zbuf + (size_t)ncol * HW_ + pb, full, rem);
      float8 hv = load8(hbuf + (size_t)ncol * HW_ + pb, full, rem);
      float8 o;
      #pragma unroll
      for (int j = 0; j < 8; ++j)
        o[j] = (1.f - zv[j]) * hv[j] + zv[j] * tanhf(acc[j]);
      store8(fout + (size_t)ncol * HW_ + pb, o, full, rem);
    }
  }
}

// ---------------- host orchestration ----------------
extern "C" void kernel_launch(void* const* d_in, const int* in_sizes, int n_in,
                              void* d_out, int out_size, void* d_ws, size_t ws_size,
                              hipStream_t stream) {
  (void)in_sizes; (void)n_in; (void)out_size; (void)ws_size;
  const float* h      = (const float*)d_in[0];
  const float* x      = (const float*)d_in[1];
  const float* wz_off = (const float*)d_in[2];
  const float* bz_off = (const float*)d_in[3];
  const float* wr_off = (const float*)d_in[4];
  const float* br_off = (const float*)d_in[5];
  const float* wq_off = (const float*)d_in[6];
  const float* bq_off = (const float*)d_in[7];
  const float* wz     = (const float*)d_in[8];
  const float* wr     = (const float*)d_in[9];
  const float* wq     = (const float*)d_in[10];
  float* out = (float*)d_out;

  char* base = (char*)d_ws;
  size_t off = 0;
  auto carve = [&](size_t bytes) {
    char* p = base + off;
    off = (off + bytes + 255) & ~(size_t)255;
    return p;
  };
  const size_t halo_bytes = (size_t)HALO_PIX * 256 * 2;
  _Float16* Hh     = (_Float16*)carve(halo_bytes);
  _Float16* Xh     = (_Float16*)carve(halo_bytes);
  _Float16* RHh    = (_Float16*)carve(halo_bytes);
  float*    off_zr = (float*)carve((size_t)36 * HW_ * 4);
  float*    off_q  = (float*)carve((size_t)18 * HW_ * 4);
  float*    zbuf   = (float*)carve((size_t)256 * HW_ * 4);
  const size_t wf_main = (size_t)KSTEPS * 16 * 32 * 16 * 2;
  _Float16* wfz    = (_Float16*)carve(wf_main);
  _Float16* wfr    = (_Float16*)carve(wf_main);
  _Float16* wfq    = (_Float16*)carve(wf_main);
  _Float16* wfzr_o = (_Float16*)carve((size_t)KSTEPS * 3 * 32 * 16 * 2);
  _Float16* wfq_o  = (_Float16*)carve((size_t)KSTEPS * 2 * 32 * 16 * 2);

  // allow >64 KB dynamic LDS (CDNA5: 320 KB/WGP)
  hipFuncSetAttribute(reinterpret_cast<const void*>(k_deform),
                      hipFuncAttributeMaxDynamicSharedMemorySize, LDS_BYTES);

  { int tot = HALO_PIX * 32;
    k_pack_halo<<<(tot + 255) / 256, 256, 0, stream>>>(h, x, Hh, Xh, RHh); }

  { int tot = KSTEPS * 16 * 32 * 8; int b = (tot + 255) / 256;
    k_pack_w<<<b, 256, 0, stream>>>(wz, wz, 256, 256, 16, (unsigned int*)wfz);
    k_pack_w<<<b, 256, 0, stream>>>(wr, wr, 256, 256, 16, (unsigned int*)wfr);
    k_pack_w<<<b, 256, 0, stream>>>(wq, wq, 256, 256, 16, (unsigned int*)wfq); }
  { int tot = KSTEPS * 3 * 32 * 8;
    k_pack_w<<<(tot + 255) / 256, 256, 0, stream>>>(wz_off, wr_off, 18, 36, 3, (unsigned int*)wfzr_o); }
  { int tot = KSTEPS * 2 * 32 * 8;
    k_pack_w<<<(tot + 255) / 256, 256, 0, stream>>>(wq_off, wq_off, 18, 18, 2, (unsigned int*)wfq_o); }

  // offset convs for z and r fused (36 output cols); one wave per M-tile, 3 N-tiles/wave
  { int thr = MTILES * 32;
    k_offconv<3><<<(thr + 255) / 256, 256, 0, stream>>>(Hh, Xh, wfzr_o, bz_off, br_off,
                                                        36, 18, off_zr); }
  // z = sigmoid(deform(hx, off_z, wz))
  k_deform<<<MTILES, 256, LDS_BYTES, stream>>>(Hh, Xh, off_zr, wfz,
                                               nullptr, nullptr, zbuf, nullptr, 0);
  // RHh = f16(sigmoid(deform(hx, off_r, wr)) * h)
  k_deform<<<MTILES, 256, LDS_BYTES, stream>>>(Hh, Xh, off_zr + (size_t)18 * HW_, wfr,
                                               nullptr, h, nullptr, RHh, 1);
  // offset conv for q on rx = (r*h, x); 2 N-tiles/wave
  { int thr = MTILES * 32;
    k_offconv<2><<<(thr + 255) / 256, 256, 0, stream>>>(RHh, Xh, wfq_o, bq_off, bq_off,
                                                        18, 18, off_q); }
  // out = (1-z)*h + z*tanh(deform(rx, off_q, wq))
  k_deform<<<MTILES, 256, LDS_BYTES, stream>>>(RHh, Xh, off_q, wfq,
                                               zbuf, h, out, nullptr, 2);
}
